// AttentionLSTM_4990751998520
// MI455X (gfx1250) — matrix operand order repaired and checked by reading, not compiled, not run on hardware
//
#include <hip/hip_runtime.h>

// ---------------------------------------------------------------------------
// Masked-LSTM final-state, MI455X (gfx1250), wave32 + WMMA f16->f32.
//
// gates = [x_t | h] @ [W_ih | W_hh]^T + bias,   K = 512 + 1024 = 1536
// Weights repacked so column' = 4*j + gate (i,f,g,o) => one 16-wide N tile
// holds all 4 gates for 4 hidden units; the owning wave does the nonlinear
// update locally.  Persistent kernel, grid barrier per step.
// A-tile (32x1536 f16) staged once per WG into LDS with
// global_load_async_to_lds_b128 (ASYNCcnt); fragments read back as
// ds_load_b128.  Weights stay L2-resident (12 MB << 192 MB).
// ---------------------------------------------------------------------------

typedef __attribute__((ext_vector_type(16))) _Float16 v16h;
typedef __attribute__((ext_vector_type(8)))  float    v8f;

#define Bsz   32
#define Tlen  2048
#define Din   512
#define Hdim  1024
#define KTOT  1536          // Din + Hdim
#define NCOL  4096          // 4*Hdim
#define PWG   64            // persistent workgroups
#define PWAVES 4            // waves per workgroup (64*4 = 256 = NCOL/16 tiles)
#define PTHREADS 128
#define ASTRIDE 1544        // A row stride in f16 (+16B pad: banks shift by 4/row)
#define SMEM_A_BYTES ((size_t)Bsz * ASTRIDE * 2)        // 98816
#define SMEM_GL_OFF  SMEM_A_BYTES
#define SMEM_BYTES   (SMEM_A_BYTES + (size_t)PWAVES * Bsz * 16 * 4)  // +8KB

// workspace layout (bytes)
#define OFF_CTR   ((size_t)0)
#define OFF_BIAS  ((size_t)256)                            // f32[4096]
#define OFF_H16   (OFF_BIAS + (size_t)NCOL*4)              // f16[32][1024]
#define OFF_H32   (OFF_H16 + (size_t)Bsz*Hdim*2)           // f32[32][1024]
#define OFF_C32   (OFF_H32 + (size_t)Bsz*Hdim*4)           // f32[32][1024]
#define OFF_WP    (OFF_C32 + (size_t)Bsz*Hdim*4)           // f16[4096][1536]
#define OFF_XH    (OFF_WP + (size_t)NCOL*KTOT*2)           // f16[32][2048][512]

union Frag { v16h h; uint4 q[2]; };

// ---------------------------------------------------------------------------
// setup kernels
// ---------------------------------------------------------------------------
__global__ void lstm_init(const float* __restrict__ h0, const float* __restrict__ c0,
                          const float* __restrict__ b_ih, const float* __restrict__ b_hh,
                          char* __restrict__ ws) {
  int tid = blockIdx.x * blockDim.x + threadIdx.x;
  if (tid == 0) *(unsigned*)(ws + OFF_CTR) = 0u;           // re-arm grid barrier
  if (tid < NCOL) {                                        // interleaved bias
    int j = tid >> 2, g = tid & 3;
    int row = g * Hdim + j;
    ((float*)(ws + OFF_BIAS))[tid] = b_ih[row] + b_hh[row];
  }
  if (tid < Bsz * Hdim) {
    float h = h0[tid], c = c0[tid];
    ((_Float16*)(ws + OFF_H16))[tid] = (_Float16)h;
    ((float*)(ws + OFF_H32))[tid] = h;
    ((float*)(ws + OFF_C32))[tid] = c;
  }
}

__global__ void pack_w(const float* __restrict__ W_ih, const float* __restrict__ W_hh,
                       char* __restrict__ ws) {
  size_t idx = (size_t)blockIdx.x * blockDim.x + threadIdx.x;
  if (idx >= (size_t)NCOL * KTOT) return;
  int colp = (int)(idx / KTOT);
  int k    = (int)(idx % KTOT);
  int j = colp >> 2, g = colp & 3;
  int row = g * Hdim + j;                                  // source gate row
  float v = (k < Din) ? W_ih[(size_t)row * Din + k]
                      : W_hh[(size_t)row * Hdim + (k - Din)];
  ((_Float16*)(ws + OFF_WP))[idx] = (_Float16)v;
}

__global__ void cvt_x(const float* __restrict__ x, char* __restrict__ ws) {
  size_t t = (size_t)blockIdx.x * blockDim.x + threadIdx.x;
  if (t >= (size_t)Bsz * Tlen * Din / 4) return;
  const float4 v = ((const float4*)x)[t];
  _Float16* xh = (_Float16*)(ws + OFF_XH) + t * 4;
  xh[0] = (_Float16)v.x; xh[1] = (_Float16)v.y;
  xh[2] = (_Float16)v.z; xh[3] = (_Float16)v.w;
}

// ---------------------------------------------------------------------------
// persistent recurrent scan
// ---------------------------------------------------------------------------
__device__ __forceinline__ float sigm(float v) { return 1.0f / (1.0f + __expf(-v)); }
// branch-free tanh, exact saturation at +/-inf via sigmoid saturation
__device__ __forceinline__ float tanh_fast(float v) { return 2.0f * sigm(2.0f * v) - 1.0f; }

__global__ __launch_bounds__(PTHREADS, 1)
void lstm_scan(const int* __restrict__ lengths, char* __restrict__ ws) {
  extern __shared__ char smem[];                 // dynamic LDS, starts at offset 0
  _Float16* As = (_Float16*)smem;                // [32][ASTRIDE] staged [x_t | h]
  float*    gl = (float*)(smem + SMEM_GL_OFF);   // [PWAVES][32*16] gate shuffle

  const int tid  = threadIdx.x;
  const int wave = tid >> 5;
  const int lane = tid & 31;
  const int half = lane >> 4;
  const int l16  = lane & 15;
  const int tile = blockIdx.x * PWAVES + wave;             // 0..255 N-tiles
  const int colp = tile * 16 + l16;                        // this lane's gate col'

  const float*    biasp = (const float*)(ws + OFF_BIAS);
  _Float16*       h16   = (_Float16*)(ws + OFF_H16);
  float*          h32   = (float*)(ws + OFF_H32);
  float*          c32   = (float*)(ws + OFF_C32);
  const _Float16* Wp    = (const _Float16*)(ws + OFF_WP);
  const _Float16* xh    = (const _Float16*)(ws + OFF_XH);
  unsigned*       ctr   = (unsigned*)(ws + OFF_CTR);

  const float bias = biasp[colp];
  const uint4* wq  = (const uint4*)(Wp + (size_t)colp * KTOT);   // 16B-aligned
  const unsigned lenB = (unsigned)lengths[lane];                 // lane == batch b

  for (int step = 0; step < Tlen; ++step) {
    // ---- stage A = [x_t | h] into LDS via async global->LDS DMA ----
    // 32 rows x 192 16B-chunks = 6144 chunks; 48 per thread
#pragma unroll 4
    for (int it = 0; it < (Bsz * (KTOT / 8)) / PTHREADS; ++it) {
      const int c   = it * PTHREADS + tid;        // 0..6143
      const int row = c / (KTOT / 8);
      const int col = (c % (KTOT / 8)) * 8;       // f16 column
      const _Float16* src = (col < Din)
          ? xh  + (size_t)row * (Tlen * Din) + (size_t)step * Din + col
          : h16 + (size_t)row * Hdim + (col - Din);
      const unsigned ldsb = (unsigned)(row * (ASTRIDE * 2) + col * 2);
      asm volatile("global_load_async_to_lds_b128 %0, %1, off"
                   :: "v"(ldsb), "v"((unsigned long long)(uintptr_t)src)
                   : "memory");
    }
    asm volatile("s_wait_asynccnt 0x0" ::: "memory");
    __syncthreads();

    // prefetch next step's x slice (global_prefetch_b8)
    __builtin_prefetch((const void*)(xh + (size_t)(step + 1) * Din), 0, 1);

    // ---- GEMM: gates(32x16 tile) += A(32x1536) * B(1536x16) ----
    v8f acc0 = {}; v8f acc1 = {};
    const uint4* arow0 = (const uint4*)(As + (size_t)l16        * ASTRIDE);
    const uint4* arow1 = (const uint4*)(As + (size_t)(l16 + 16) * ASTRIDE);
#pragma unroll 2
    for (int kk = 0; kk < KTOT / 32; ++kk) {
      const int k8 = kk * 4;                      // kb/8, kb = kk*32
      Frag fa0, fa1, fb;
      // ISA 7.12.2 layouts: two 16B runs per fragment -> ds_load_b128 x2
      fb.q[0]  = wq[k8 + half];
      fb.q[1]  = wq[k8 + half + 2];
      fa0.q[0] = arow0[k8 + half];
      fa0.q[1] = arow0[k8 + half + 2];
      fa1.q[0] = arow1[k8 + half];
      fa1.q[1] = arow1[k8 + half + 2];
      acc0 = __builtin_amdgcn_wmma_f32_16x16x32_f16(false, fa0.h, false, fb.h,
                                                    (short)0, acc0, false, false);
      acc1 = __builtin_amdgcn_wmma_f32_16x16x32_f16(false, fa1.h, false, fb.h,
                                                    (short)0, acc1, false, false);
    }

    // ---- stage gates (+bias) through LDS: D layout lane(n)=l16, m=8*half+r ----
#pragma unroll
    for (int r = 0; r < 8; ++r) {
      gl[wave * (Bsz * 16) + (8 * half + r) * 16 + l16]      = acc0[r] + bias;
      gl[wave * (Bsz * 16) + (16 + 8 * half + r) * 16 + l16] = acc1[r] + bias;
    }
    asm volatile("s_wait_dscnt 0x0" ::: "memory");   // wave-local LDS RAW

    // ---- elementwise update: lane = batch b; 4 hidden units per wave-tile ----
    const bool upd = ((unsigned)step) < lenB;
    const int  b   = lane;
#pragma unroll
    for (int jj = 0; jj < 4; ++jj) {
      const float4 g4 = *(const float4*)&gl[wave * (Bsz * 16) + b * 16 + jj * 4];
      const int    j  = tile * 4 + jj;
      const size_t hi = (size_t)b * Hdim + j;
      float cold = c32[hi];
      float cn = sigm(g4.y) * cold + sigm(g4.x) * tanh_fast(g4.z);
      float hn = sigm(g4.w) * tanh_fast(cn);
      if (upd) {
        c32[hi] = cn;
        h32[hi] = hn;
        h16[hi] = (_Float16)hn;
      }
    }

    // ---- grid-wide barrier (release h16/c32, acquire for next step) ----
    __threadfence();
    __syncthreads();
    if (tid == 0) {
      atomicAdd(ctr, 1u);
      const unsigned target = (unsigned)(step + 1) * gridDim.x;
      while (__hip_atomic_load(ctr, __ATOMIC_ACQUIRE, __HIP_MEMORY_SCOPE_AGENT) < target) {
        __builtin_amdgcn_s_sleep(1);
      }
    }
    __syncthreads();
    __threadfence();
  }
}

// ---------------------------------------------------------------------------
__global__ void lstm_out(const char* __restrict__ ws, float* __restrict__ out) {
  int i = blockIdx.x * blockDim.x + threadIdx.x;
  if (i < Bsz * Hdim)            out[i] = ((const float*)(ws + OFF_H32))[i];
  else if (i < 2 * Bsz * Hdim)   out[i] = ((const float*)(ws + OFF_C32))[i - Bsz * Hdim];
}

// ---------------------------------------------------------------------------
extern "C" void kernel_launch(void* const* d_in, const int* in_sizes, int n_in,
                              void* d_out, int out_size, void* d_ws, size_t ws_size,
                              hipStream_t stream) {
  const float* x       = (const float*)d_in[0];
  const int*   lengths = (const int*)  d_in[1];
  const float* h0      = (const float*)d_in[2];
  const float* c0      = (const float*)d_in[3];
  const float* W_ih    = (const float*)d_in[4];
  const float* W_hh    = (const float*)d_in[5];
  const float* b_ih    = (const float*)d_in[6];
  const float* b_hh    = (const float*)d_in[7];
  char*  ws  = (char*)d_ws;
  float* out = (float*)d_out;

  lstm_init<<<(Bsz * Hdim + 255) / 256, 256, 0, stream>>>(h0, c0, b_ih, b_hh, ws);

  {
    size_t n = (size_t)NCOL * KTOT;
    pack_w<<<(unsigned)((n + 255) / 256), 256, 0, stream>>>(W_ih, W_hh, ws);
  }
  {
    size_t n = (size_t)Bsz * Tlen * Din / 4;
    cvt_x<<<(unsigned)((n + 255) / 256), 256, 0, stream>>>(x, ws);
  }

  lstm_scan<<<PWG, PTHREADS, (unsigned)SMEM_BYTES, stream>>>(lengths, ws);

  lstm_out<<<(2 * Bsz * Hdim + 255) / 256, 256, 0, stream>>>(ws, out);
}
